// mutualPoolingDir_45234595562201
// MI455X (gfx1250) — matrix operand order repaired and checked by reading, not compiled, not run on hardware
//
#include <hip/hip_runtime.h>
#include <hip/hip_bf16.h>

typedef __attribute__((ext_vector_type(16))) _Float16 v16h;
typedef __attribute__((ext_vector_type(8)))  float    v8f;

constexpr int G = 64, L = 100, D = 128, P = 64;
constexpr int LDK = D + 4;      // padded LDS row stride (floats) -> bank-conflict free
constexpr int PCHUNK = 16;      // queries handled per block
constexpr int NTHREADS = 256;   // 8 waves (wave32)

__global__ __launch_bounds__(NTHREADS)
void mutual_pool_fused_kernel(const float* __restrict__ gallery_value,
                              const float* __restrict__ gallery_base,
                              const float* __restrict__ querys,
                              const float* __restrict__ Wk,
                              const float* __restrict__ bk,
                              float* __restrict__ out)
{
    extern __shared__ float smem[];
    float* Ks = smem;                 // [L][LDK]  projected keys (f32)
    float* Vs = smem + L * LDK;       // [L][LDK]  gallery_value tile
    float* qs = smem + 2 * L * LDK;   // [PCHUNK][D]

    const int g    = blockIdx.x;
    const int pc   = blockIdx.y;
    const int tid  = threadIdx.x;
    const int lane = tid & 31;
    const int wave = tid >> 5;        // 0..7 -> this wave's 16-wide N tile
    const int half = lane >> 4;       // lane group (0: lanes 0-15, 1: lanes 16-31)
    const int l16  = lane & 15;

    // ---------------- Phase 1: K[g] = gallery_base[g] @ Wk^T + bk  (WMMA) ---------------
    // B matrix for WMMA is [Kdim=128 x N=16]: B[d][e] = Wk[e][d]. Hoist all 4 k-chunks.
    const int e = wave * 16 + l16;            // output feature column this lane owns
    const float* wrow = Wk + e * D;           // Wk row e is contiguous in d
    v16h bfrag[4];
    #pragma unroll
    for (int kc = 0; kc < 4; ++kc) {
        const int kb = kc * 32 + 8 * half;    // k = 2(v&3)+8*half+16*(v>>2), same as A
        #pragma unroll
        for (int v = 0; v < 8; ++v) {
            const float2 f = *(const float2*)(wrow + kb + 2 * (v & 3) + 16 * (v >> 2));
            bfrag[kc][2 * v]     = (_Float16)f.x;
            bfrag[kc][2 * v + 1] = (_Float16)f.y;
        }
    }
    const float bias = bk[e];

    #pragma unroll
    for (int mt = 0; mt < 7; ++mt) {          // 7 x 16 rows covers L=100 (clamped)
        const int lrow = mt * 16 + l16;
        const int lcl  = lrow < L ? lrow : (L - 1);   // clamp: garbage rows never stored
        const float* arow = gallery_base + (g * L + lcl) * D;
        v8f c = {};
        #pragma unroll
        for (int kc = 0; kc < 4; ++kc) {
            v16h a;
            #pragma unroll
            for (int v = 0; v < 8; ++v) {     // ISA A 16x32 f16: k = 2(v&3)+8*half+16*(v>>2)
                const int kp = kc * 32 + 2 * (v & 3) + 8 * half + 16 * (v >> 2);
                const float2 f = *(const float2*)(arow + kp);
                a[2 * v]     = (_Float16)f.x;
                a[2 * v + 1] = (_Float16)f.y;
            }
            c = __builtin_amdgcn_wmma_f32_16x16x32_f16(false, a, false, bfrag[kc],
                                                       (short)0, c, false, false);
        }
        #pragma unroll
        for (int r = 0; r < 8; ++r) {         // C/D layout: row = r + 8*half, col = e
            const int l = mt * 16 + r + 8 * half;
            if (l < L) Ks[l * LDK + e] = c[r] + bias;
        }
    }

    // ---------------- Stage V tile and this block's query chunk into LDS ----------------
    for (int i = tid; i < L * D; i += NTHREADS) {
        const int l = i >> 7, d = i & (D - 1);
        Vs[l * LDK + d] = gallery_value[(g * L + l) * D + d];
    }
    for (int i = tid; i < PCHUNK * D; i += NTHREADS) {
        qs[i] = querys[pc * PCHUNK * D + i];
    }
    __syncthreads();

    // ------------- Phase 2: per-(p,d) stable softmax over l, pool gallery_value ---------
    #pragma unroll
    for (int t = 0; t < (PCHUNK * D) / NTHREADS; ++t) {   // 8 tasks / thread
        const int id = tid + t * NTHREADS;
        const int pl = id >> 7;               // local query index
        const int d  = id & (D - 1);
        const float q = qs[pl * D + d];

        float m = -3.402823466e+38f;
        #pragma unroll 4
        for (int l = 0; l < L; ++l)
            m = fmaxf(m, q * Ks[l * LDK + d]);

        float s = 0.f, acc = 0.f;
        #pragma unroll 4
        for (int l = 0; l < L; ++l) {
            const float w = __expf(q * Ks[l * LDK + d] - m);
            s   += w;
            acc += w * Vs[l * LDK + d];
        }
        const int p = pc * PCHUNK + pl;
        out[(p * G + g) * D + d] = acc / s;   // out is [P, G, D]
    }
}

extern "C" void kernel_launch(void* const* d_in, const int* in_sizes, int n_in,
                              void* d_out, int out_size, void* d_ws, size_t ws_size,
                              hipStream_t stream) {
    (void)in_sizes; (void)n_in; (void)out_size; (void)d_ws; (void)ws_size;
    const float* gallery_value = (const float*)d_in[0];
    const float* gallery_base  = (const float*)d_in[1];
    const float* querys        = (const float*)d_in[2];
    const float* Wk            = (const float*)d_in[3];
    const float* bk            = (const float*)d_in[4];
    float* out = (float*)d_out;

    const size_t smem_bytes = (size_t)(2 * L * LDK + PCHUNK * D) * sizeof(float); // 113,792 B
    dim3 grid(G, P / PCHUNK);   // (64, 4) = 256 blocks ~= 1 per WGP
    mutual_pool_fused_kernel<<<grid, NTHREADS, smem_bytes, stream>>>(
        gallery_value, gallery_base, querys, Wk, bk, out);
}
